// MultiHeadAttention_17721035063724
// MI455X (gfx1250) — matrix operand order repaired and checked by reading, not compiled, not run on hardware
//
#include <hip/hip_runtime.h>
#include <hip/hip_bf16.h>

// MHA forward (b=2, s=2048, dim=1024, 16 heads, hd=64) for gfx1250.
// bf16 WMMA everywhere; TDM tensor_load_to_lds double-buffered staging;
// batched ds_swizzle cross-lane softmax reductions (branchless).

#define BATCH 2
#define SEQ   2048
#define DIM   1024
#define NHEAD 16
#define HDIM  64

typedef __attribute__((ext_vector_type(4)))  int            v4i;
typedef __attribute__((ext_vector_type(8)))  int            v8i;
typedef __attribute__((ext_vector_type(4)))  unsigned int   u32x4;
typedef __attribute__((ext_vector_type(4)))  float          v4f;
typedef __attribute__((ext_vector_type(4)))  unsigned short u16x4;
typedef __attribute__((ext_vector_type(8)))  float          v8f;
typedef __attribute__((ext_vector_type(16))) __bf16         v16bf;

union Frag16 { v16bf v; u32x4 q[2]; };

#if __has_builtin(__builtin_amdgcn_global_load_async_to_lds_b128)
#define HAS_ASYNC_LDS 1
#else
#define HAS_ASYNC_LDS 0
#endif

#if __has_builtin(__builtin_amdgcn_s_wait_tensorcnt) && \
    (__has_builtin(__builtin_amdgcn_tensor_load_to_lds_d2) || \
     __has_builtin(__builtin_amdgcn_tensor_load_to_lds))
#define HAS_TDM 1
#else
#define HAS_TDM 0
#endif

__device__ __forceinline__ unsigned short f2bf(float f) {
  union { float f; unsigned int u; } cv; cv.f = f;
  unsigned int u = cv.u;
  unsigned int r = u + 0x7FFFu + ((u >> 16) & 1u);   // round-to-nearest-even
  return (unsigned short)(r >> 16);
}

// copy 16 bytes global -> LDS; async (ASYNCcnt) when available
__device__ __forceinline__ void stage16(const unsigned short *g, unsigned short *l) {
#if HAS_ASYNC_LDS
  __builtin_amdgcn_global_load_async_to_lds_b128(
      (__attribute__((address_space(1))) v4i *)(size_t)g,
      (__attribute__((address_space(3))) v4i *)(unsigned int)(size_t)l, 0, 0);
#else
  *(u32x4 *)l = *(const u32x4 *)g;
#endif
}

#if HAS_TDM
// One TDM descriptor: 2-D tile (tx elems per row, ty rows) of 2-byte elements,
// global row stride `stride` (elems), LDS rows padded per pint/pamt codes
// (pad_interval: 2^(c+1) DWORDs between pads; pad_amount: c+1 DWORDs).
__device__ __forceinline__ void tdm2d(const unsigned short *g, unsigned short *l,
                                      int tx, int ty, long long stride,
                                      int pint, int pamt) {
  unsigned long long ga = (unsigned long long)(size_t)g;
  unsigned int la = (unsigned int)(size_t)l;
  u32x4 g0;
  g0[0] = 1u;                                            // count=1 (valid D#)
  g0[1] = la;                                            // lds_addr (bytes)
  g0[2] = (unsigned int)ga;                              // global_addr[31:0]
  g0[3] = (unsigned int)((ga >> 32) & 0x1FFFFFFull) | (2u << 30);  // +type=2
  const unsigned int T0 = 1u << 30, T1 = 1u << 30;       // huge dims: no clamp
  v8i g1;
  g1[0] = (int)((1u << 16) |                             // data_size = 2 bytes
                (1u << 20) |                             // pad_enable
                ((unsigned int)pint << 22) |             // pad_interval
                ((unsigned int)pamt << 25));             // pad_amount
  g1[1] = (int)((T0 & 0xFFFFu) << 16);                   // tensor_dim0 lo16
  g1[2] = (int)(((T0 >> 16) & 0xFFFFu) | ((T1 & 0xFFFFu) << 16));
  g1[3] = (int)(((T1 >> 16) & 0xFFFFu) | ((unsigned int)tx << 16)); // tile_dim0
  g1[4] = ty;                                            // tile_dim1 (dim2=0)
  g1[5] = (int)(unsigned int)stride;                     // dim0 stride lo32
  g1[6] = (int)((unsigned int)(stride >> 32) & 0xFFFFu); // stride hi16
  g1[7] = 0;
#if __has_builtin(__builtin_amdgcn_tensor_load_to_lds_d2)
  __builtin_amdgcn_tensor_load_to_lds_d2(g0, g1, 0);
#else
  v4i z4 = (v4i)0;
  v8i z8 = (v8i)0;
  __builtin_amdgcn_tensor_load_to_lds(g0, g1, z4, z4, z8, 0);
#endif
}
#endif

// wait for this wave's outstanding staging ops (TDM or async)
__device__ __forceinline__ void wait_stage(bool issuer) {
#if HAS_TDM
  if (issuer) __builtin_amdgcn_s_wait_tensorcnt(0);
#elif HAS_ASYNC_LDS
  (void)issuer;
#if __has_builtin(__builtin_amdgcn_s_wait_asynccnt)
  __builtin_amdgcn_s_wait_asynccnt(0);
#else
  asm volatile("s_wait_asynccnt 0x0" ::: "memory");
#endif
#else
  (void)issuer;
#endif
}

__device__ __forceinline__ v8f wmma_bf16(const Frag16 &a, const Frag16 &b, v8f c) {
  return __builtin_amdgcn_wmma_f32_16x16x32_bf16(false, a.v, false, b.v,
                                                 (short)0, c, false, false);
}

__device__ __forceinline__ u32x4 lds16(const unsigned short *p) {
  return *(const u32x4 *)p;
}

// lane-xor shuffle within 32 lanes: single ds_swizzle (group-of-32 mode)
template <int MASK>
__device__ __forceinline__ float swz_xor(float v) {
  int i = __builtin_amdgcn_ds_swizzle(__builtin_bit_cast(int, v),
                                      (MASK << 10) | 0x1F);
  return __builtin_bit_cast(float, i);
}

// batched 16-lane reductions over 8 independent rows: issue 8 swizzles per
// step so the LDS pipeline stays full (partial dscnt waits instead of 0)
template <int MASK>
__device__ __forceinline__ void stepmax8(float (&v)[8]) {
  float t[8];
#pragma unroll
  for (int r = 0; r < 8; ++r) t[r] = swz_xor<MASK>(v[r]);
#pragma unroll
  for (int r = 0; r < 8; ++r) v[r] = fmaxf(v[r], t[r]);
}
template <int MASK>
__device__ __forceinline__ void stepsum8(float (&v)[8]) {
  float t[8];
#pragma unroll
  for (int r = 0; r < 8; ++r) t[r] = swz_xor<MASK>(v[r]);
#pragma unroll
  for (int r = 0; r < 8; ++r) v[r] += t[r];
}
__device__ __forceinline__ void xmax16x8(float (&v)[8]) {
  stepmax8<1>(v); stepmax8<2>(v); stepmax8<4>(v); stepmax8<8>(v);
}
__device__ __forceinline__ void xsum16x8(float (&v)[8]) {
  stepsum8<1>(v); stepsum8<2>(v); stepsum8<4>(v); stepsum8<8>(v);
}

// ---------------------------------------------------------------- fp32 -> bf16
__global__ void __launch_bounds__(256) cvt_bf16(const float *__restrict__ src,
                                                unsigned short *__restrict__ dst,
                                                int n4) {   // n4 = n/4
  int i = blockIdx.x * 256 + threadIdx.x;
  int stride = gridDim.x * 256;
  for (; i < n4; i += stride) {
    v4f f = *(const v4f *)(src + (size_t)i * 4);
    u16x4 o;
#pragma unroll
    for (int j = 0; j < 4; ++j) o[j] = f2bf(f[j]);
    *(u16x4 *)(dst + (size_t)i * 4) = o;
  }
}

// ------------------------------------------------- 128x128 WMMA GEMM, K = 1024
// C[m][e] = sum_k A[m][k] * W[e][k]; K-step 64, double-buffered TDM staging.
// MODE 0: scatter into q (scaled), k, v^T head-major bf16 buffers.
// MODE 1: plain f32 store (out projection).
template <int MODE>
__global__ void __launch_bounds__(256) gemm_wmma(
    const unsigned short *__restrict__ A, const unsigned short *__restrict__ W,
    unsigned short *__restrict__ qb, unsigned short *__restrict__ kb,
    unsigned short *__restrict__ vtb, float *__restrict__ outF) {
  __shared__ unsigned short sA[2][128 * 72];   // 128 rows x 64 K (pad 72)
  __shared__ unsigned short sW[2][128 * 72];

  const int tid  = threadIdx.x;
  const int w    = tid >> 5;
  const int lane = tid & 31;
  const int lm   = lane & 15;
  const int h    = lane >> 4;
  const bool issuer = (w == 0);   // wave-uniform

  const int m0 = blockIdx.x * 128;
  const int n0 = blockIdx.y * 128;

  v8f acc[8];
#pragma unroll
  for (int f = 0; f < 8; ++f) acc[f] = (v8f)0.0f;

  // stage tile kt into buffer bi
  auto stage_tiles = [&](int bi, int kt) {
#if HAS_TDM
    if (issuer) {   // one DMA per matrix tile: 128 rows x 64 elems, pad to 72
      tdm2d(A + (size_t)m0 * 1024 + kt * 64, &sA[bi][0], 64, 128, 1024, 4, 3);
      tdm2d(W + (size_t)n0 * 1024 + kt * 64, &sW[bi][0], 64, 128, 1024, 4, 3);
    }
#else
#pragma unroll
    for (int i = 0; i < 4; ++i) {
      int c = tid + 256 * i;
      int r = c >> 3, seg = c & 7;
      stage16(A + (size_t)(m0 + r) * 1024 + kt * 64 + seg * 8,
              &sA[bi][r * 72 + seg * 8]);
      stage16(W + (size_t)(n0 + r) * 1024 + kt * 64 + seg * 8,
              &sW[bi][r * 72 + seg * 8]);
    }
#endif
  };

  stage_tiles(0, 0);
#pragma unroll 2
  for (int kt = 0; kt < 1024 / 64; ++kt) {
    const int cur = kt & 1;
    wait_stage(issuer);
    __syncthreads();
    if (kt + 1 < 1024 / 64) stage_tiles(cur ^ 1, kt + 1);

    const unsigned short *bA = sA[cur];
    const unsigned short *bW = sW[cur];
#pragma unroll
    for (int ks = 0; ks < 2; ++ks) {
      Frag16 a;   // A frag: K chunks at 8h and 8h+16 (ISA A-matrix layout)
      const unsigned short *row = &bA[(16 * w + lm) * 72 + ks * 32];
      a.q[0] = lds16(row + 8 * h);
      a.q[1] = lds16(row + 8 * h + 16);
#pragma unroll
      for (int f = 0; f < 8; ++f) {
        Frag16 b;  // B frag: K chunks at 8h and 8h+16 (matches A layout)
        const unsigned short *rw = &bW[(16 * f + lm) * 72 + ks * 32 + 8 * h];
        b.q[0] = lds16(rw);
        b.q[1] = lds16(rw + 16);
        acc[f] = wmma_bf16(a, b, acc[f]);
      }
    }
  }

  // epilogue: C layout m = r + 8*half, n = lane&15
#pragma unroll
  for (int f = 0; f < 8; ++f) {
#pragma unroll
    for (int r = 0; r < 8; ++r) {
      float val = acc[f][r];
      int row = m0 + 16 * w + r + 8 * h;
      int e   = n0 + 16 * f + lm;
      if (MODE == 0) {
        int which = e >> 10;                 // 0=q 1=k 2=v
        int hh = (e >> 6) & 15, d = e & 63;
        int bi = row >> 11, si = row & 2047;
        if (which == 0)
          qb[((bi * NHEAD + hh) * SEQ + si) * HDIM + d] = f2bf(val * 0.125f);
        else if (which == 1)
          kb[((bi * NHEAD + hh) * SEQ + si) * HDIM + d] = f2bf(val);
        else
          vtb[((bi * NHEAD + hh) * HDIM + d) * SEQ + si] = f2bf(val);
      } else {
        outF[(size_t)row * 1024 + e] = val;
      }
    }
  }
}

// --------------------------------------------------------- flash attention
// grid: (s/128, n_heads, batch); block: 256 (8 waves, each owns 16 q rows)
__global__ void __launch_bounds__(256) attn_wmma(
    const unsigned short *__restrict__ qg, const unsigned short *__restrict__ kg,
    const unsigned short *__restrict__ vtg, const unsigned char *__restrict__ msk,
    unsigned short *__restrict__ ao) {
  __shared__ unsigned short sQ[128 * 72];       // 128 q-rows x 64 d (pad 72)
  __shared__ unsigned short sK[2][32 * 72];     // 32 keys x 64 d (pad 72)
  __shared__ unsigned short sVt[2][64 * 40];    // 64 d x 32 keys (pad 40)
  __shared__ unsigned short sP[8][16 * 40];     // per-wave P tile 16x32 (pad 40)

  const int tid  = threadIdx.x;
  const int w    = tid >> 5;
  const int lane = tid & 31;
  const int lm   = lane & 15;
  const int h    = lane >> 4;
  const bool issuer = (w == 0);   // wave-uniform

  const int bh  = blockIdx.z * NHEAD + blockIdx.y;
  const int qb0 = blockIdx.x * 128;
  const unsigned short *qbase = qg  + (size_t)bh * SEQ * HDIM;
  const unsigned short *kbase = kg  + (size_t)bh * SEQ * HDIM;
  const unsigned short *vbase = vtg + (size_t)bh * HDIM * SEQ;
  const unsigned char  *mbase = msk + (size_t)blockIdx.z * SEQ;

  auto stage_kv = [&](int bi, int jt) {
#if HAS_TDM
    if (issuer) {
      // K tile: 32 keys x 64 d, global row stride 64, LDS rows padded to 72
      tdm2d(kbase + (size_t)jt * 32 * HDIM, &sK[bi][0], 64, 32, 64, 4, 3);
      // V^T tile: 64 d x 32 keys, global row stride SEQ, LDS rows padded to 40
      tdm2d(vbase + jt * 32, &sVt[bi][0], 32, 64, SEQ, 3, 3);
    }
#else
    int r = tid >> 3, seg = tid & 7;
    stage16(kbase + (size_t)(jt * 32 + r) * HDIM + seg * 8,
            &sK[bi][r * 72 + seg * 8]);
    int rv = tid >> 2, sv = tid & 3;
    stage16(vbase + (size_t)rv * SEQ + jt * 32 + sv * 8,
            &sVt[bi][rv * 40 + sv * 8]);
#endif
  };

  // stage Q block + first K/V tile
#if HAS_TDM
  if (issuer)
    tdm2d(qbase + (size_t)qb0 * HDIM, &sQ[0], 64, 128, 64, 4, 3);
#else
#pragma unroll
  for (int i = 0; i < 4; ++i) {
    int c = tid + 256 * i;
    int r = c >> 3, seg = c & 7;
    stage16(qbase + (size_t)(qb0 + r) * HDIM + seg * 8, &sQ[r * 72 + seg * 8]);
  }
#endif
  stage_kv(0, 0);
  wait_stage(issuer);
  __syncthreads();

  Frag16 aQ[2];   // Q strip 16x64 as two 16x32 A-frags
#pragma unroll
  for (int ks = 0; ks < 2; ++ks) {
    const unsigned short *row = &sQ[(16 * w + lm) * 72 + ks * 32];
    aQ[ks].q[0] = lds16(row + 8 * h);
    aQ[ks].q[1] = lds16(row + 8 * h + 16);
  }

  float rmax[8], rsum[8];
  v8f o[4];
#pragma unroll
  for (int r = 0; r < 8; ++r) { rmax[r] = -INFINITY; rsum[r] = 0.0f; }
#pragma unroll
  for (int f = 0; f < 4; ++f) o[f] = (v8f)0.0f;

  const int nkt = (qb0 + 128) / 32;   // causal: always a multiple of 4
#pragma unroll 2
  for (int jt = 0; jt < nkt; ++jt) {
    const int cur = jt & 1;
    if (jt) {                 // tile 0 already waited/synced above
      wait_stage(issuer);
      __syncthreads();
    }
    if (jt + 1 < nkt) stage_kv(cur ^ 1, jt + 1);

    // S = Q * K^T  (16 q x 32 keys per wave, two 16-key frags)
    v8f s0 = (v8f)0.0f, s1 = (v8f)0.0f;
#pragma unroll
    for (int ks = 0; ks < 2; ++ks) {
      Frag16 b0, b1;
      const unsigned short *r0 = &sK[cur][lm * 72 + ks * 32 + 8 * h];
      const unsigned short *r1 = &sK[cur][(16 + lm) * 72 + ks * 32 + 8 * h];
      b0.q[0] = lds16(r0); b0.q[1] = lds16(r0 + 16);
      b1.q[0] = lds16(r1); b1.q[1] = lds16(r1 + 16);
      s0 = wmma_bf16(aQ[ks], b0, s0);
      s1 = wmma_bf16(aQ[ks], b1, s1);
    }

    // masks (row m = r + 8*half; col = key)
    const int kc0 = jt * 32 + lm, kc1 = kc0 + 16;
    const bool ms0 = mbase[kc0] != 0, ms1 = mbase[kc1] != 0;
    const int qrow0 = qb0 + 16 * w + 8 * h;
    float v0a[8], v1a[8], sv[8];
#pragma unroll
    for (int r = 0; r < 8; ++r) {
      int qr = qrow0 + r;
      v0a[r] = (ms0 || kc0 > qr) ? -INFINITY : s0[r];
      v1a[r] = (ms1 || kc1 > qr) ? -INFINITY : s1[r];
      sv[r] = fmaxf(v0a[r], v1a[r]);
    }
    xmax16x8(sv);   // per-row tile max (batched swizzle reduction)

    // branchless online softmax: clamp keeps exp() well-defined when a row
    // has no live keys yet (everything stays 0; final epilogue guards sum==0)
    float ps[8];
#pragma unroll
    for (int r = 0; r < 8; ++r) {
      float nm  = fmaxf(rmax[r], sv[r]);
      float nmc = fmaxf(nm, -3.0e38f);
      float corr = __expf(rmax[r] - nmc);
      float p0 = __expf(v0a[r] - nmc);
      float p1 = __expf(v1a[r] - nmc);
      rmax[r] = nm;
      rsum[r] *= corr;
      ps[r] = p0 + p1;
#pragma unroll
      for (int f = 0; f < 4; ++f) o[f][r] *= corr;
      sP[w][(r + 8 * h) * 40 + lm]      = f2bf(p0);
      sP[w][(r + 8 * h) * 40 + 16 + lm] = f2bf(p1);
    }
    xsum16x8(ps);   // per-row tile sum (batched swizzle reduction)
#pragma unroll
    for (int r = 0; r < 8; ++r) rsum[r] += ps[r];

    // O += P * V  (contraction over 32 keys; wave-private LDS round-trip for P)
    Frag16 aP;
    {
      const unsigned short *row = &sP[w][lm * 40];
      aP.q[0] = lds16(row + 8 * h);
      aP.q[1] = lds16(row + 8 * h + 16);
    }
#pragma unroll
    for (int f = 0; f < 4; ++f) {
      Frag16 bV;
      const unsigned short *row = &sVt[cur][(16 * f + lm) * 40 + 8 * h];
      bV.q[0] = lds16(row);
      bV.q[1] = lds16(row + 16);
      o[f] = wmma_bf16(aP, bV, o[f]);
    }
  }

  // normalize (one v_rcp per row) + store to attn-out buffer [b][s][dim] bf16
  float rinv[8];
#pragma unroll
  for (int r = 0; r < 8; ++r)
    rinv[r] = (rsum[r] > 0.0f) ? __builtin_amdgcn_rcpf(rsum[r]) : 0.0f;
#pragma unroll
  for (int f = 0; f < 4; ++f) {
#pragma unroll
    for (int r = 0; r < 8; ++r) {
      float val = o[f][r] * rinv[r];
      int sr = qb0 + 16 * w + r + 8 * h;
      ao[((size_t)blockIdx.z * SEQ + sr) * DIM + blockIdx.y * HDIM + 16 * f + lm] =
          f2bf(val);
    }
  }
}

// ------------------------------------------------------------------- launcher
extern "C" void kernel_launch(void *const *d_in, const int *in_sizes, int n_in,
                              void *d_out, int out_size, void *d_ws,
                              size_t ws_size, hipStream_t stream) {
  (void)in_sizes; (void)n_in; (void)out_size; (void)ws_size;
  const float *x     = (const float *)d_in[0];
  const unsigned char *src_mask = (const unsigned char *)d_in[1];  // bool[b][s]
  const float *W_qkv = (const float *)d_in[2];
  const float *W_o   = (const float *)d_in[3];
  float *out = (float *)d_out;

  char *ws = (char *)d_ws;
  const size_t TOK = (size_t)BATCH * SEQ;              // 4096
  unsigned short *xb    = (unsigned short *)(ws);                       // 8 MB
  unsigned short *wqkvb = (unsigned short *)(ws + (8u << 20));          // 6 MB
  unsigned short *wob   = (unsigned short *)(ws + (14u << 20));         // 2 MB
  unsigned short *qbuf  = (unsigned short *)(ws + (16u << 20));         // 8 MB
  unsigned short *kbuf  = (unsigned short *)(ws + (24u << 20));         // 8 MB
  unsigned short *vtbuf = (unsigned short *)(ws + (32u << 20));         // 8 MB
  unsigned short *aobuf = (unsigned short *)(ws + (40u << 20));         // 8 MB

  // 1) fp32 -> bf16
  int nx = (int)(TOK * DIM) / 4;
  int nq = (3 * DIM * DIM) / 4;
  int no = (DIM * DIM) / 4;
  cvt_bf16<<<(nx + 255) / 256, 256, 0, stream>>>(x, xb, nx);
  cvt_bf16<<<(nq + 255) / 256, 256, 0, stream>>>(W_qkv, wqkvb, nq);
  cvt_bf16<<<(no + 255) / 256, 256, 0, stream>>>(W_o, wob, no);

  // 2) fused QKV projection: [4096,1024] x [3072,1024]^T, scatter q/k/v^T
  gemm_wmma<0><<<dim3(TOK / 128, (3 * DIM) / 128), 256, 0, stream>>>(
      xb, wqkvb, qbuf, kbuf, vtbuf, nullptr);

  // 3) flash attention per (q-block, head, batch)
  attn_wmma<<<dim3(SEQ / 128, NHEAD, BATCH), 256, 0, stream>>>(
      qbuf, kbuf, vtbuf, src_mask, aobuf);

  // 4) output projection: [4096,1024] x [1024,1024]^T -> f32 out
  gemm_wmma<1><<<dim3(TOK / 128, DIM / 128), 256, 0, stream>>>(
      aobuf, wob, nullptr, nullptr, nullptr, out);
}
